// GIN_5643587027248
// MI455X (gfx1250) — matrix operand order repaired and checked by reading, not compiled, not run on hardware
//
#include <hip/hip_runtime.h>
#include <hip/hip_bf16.h>

typedef __bf16 bf16t;
typedef __attribute__((ext_vector_type(16))) __bf16 v16bf;
typedef __attribute__((ext_vector_type(8)))  __bf16 v8bf;
typedef __attribute__((ext_vector_type(4)))  __bf16 v4bf;
typedef __attribute__((ext_vector_type(8)))  float  v8f;

#define N_NODES 100000
#define N_EDGES 1600000
#define HD 128
#define NL 4
#define NG 128
#define NC 10
#define BNEPS 1e-5f

#define SW_STRIDE 136   // bf16 elems; 272B rows -> 16B aligned, bank step 4 (conflict-free)
#define SA_STRIDE 40    // bf16 elems; 80B rows  -> 16B aligned, bank step 20 (conflict-free)

// ---------------------------------------------------------------- utilities
__global__ void k_zero(float* __restrict__ p, int n) {
  int stride = gridDim.x * blockDim.x;
  for (int i = blockIdx.x * blockDim.x + threadIdx.x; i < n; i += stride) p[i] = 0.0f;
}

__global__ void k_copy4(float4* __restrict__ dst, const float4* __restrict__ src, size_t n4) {
  size_t stride = (size_t)gridDim.x * blockDim.x;
  for (size_t i = (size_t)blockIdx.x * blockDim.x + threadIdx.x; i < n4; i += stride)
    dst[i] = src[i];
}

// ------------------------------------------------- edge scatter aggregation
// h (pre-initialized to x) += sum over edges: h[dst] += x[src]
// one wave covers one edge: 32 lanes x float4 = 128 features
__global__ void k_edge(const float* __restrict__ xin, const int* __restrict__ ei,
                       float* __restrict__ h) {
  const size_t total  = (size_t)N_EDGES * 32;
  const size_t stride = (size_t)gridDim.x * blockDim.x;
  for (size_t i = (size_t)blockIdx.x * blockDim.x + threadIdx.x; i < total; i += stride) {
    const int e = (int)(i >> 5);
    const int f = (int)(i & 31) * 4;
    const int s = ei[e];
    const int d = ei[N_EDGES + e];
    const float4 v = *(const float4*)&xin[(size_t)s * HD + f];
    float* dp = &h[(size_t)d * HD + f];
    atomicAdd(dp + 0, v.x);
    atomicAdd(dp + 1, v.y);
    atomicAdd(dp + 2, v.z);
    atomicAdd(dp + 3, v.w);
  }
}

// ------------------------------------------------------------- WMMA GEMM
// Branch-free slab prefetch: clamp row instead of zero-filling. OOB A rows only
// influence output rows >= n_rows, which the epilogue never stores.
__device__ __forceinline__ void load_slab(const float* __restrict__ A, int row0, int n_rows,
                                          int k0, int tid, float4* pre) {
#pragma unroll
  for (int j = 0; j < 4; ++j) {
    const int c  = tid + j * 256;       // 1024 float4 chunks cover 128x32 slab
    const int r  = c >> 3;
    const int kc = (c & 7) * 4;
    const int row = min(row0 + r, n_rows - 1);
    pre[j] = *(const float4*)&A[(size_t)row * HD + k0 + kc];
  }
}

__device__ __forceinline__ void store_slab(bf16t* __restrict__ buf, int tid, const float4* pre) {
#pragma unroll
  for (int j = 0; j < 4; ++j) {
    const int c  = tid + j * 256;
    const int r  = c >> 3;
    const int kc = (c & 7) * 4;
    v4bf pk;
    pk[0] = (bf16t)pre[j].x; pk[1] = (bf16t)pre[j].y;
    pk[2] = (bf16t)pre[j].z; pk[3] = (bf16t)pre[j].w;
    *(v4bf*)&buf[r * SA_STRIDE + kc] = pk;
  }
}

#define CAT16(lo, hi) __builtin_shufflevector(lo, hi, 0,1,2,3,4,5,6,7,8,9,10,11,12,13,14,15)

// D[n_rows x 128] = act(A[n_rows x 128] @ W[128 x 128] + bias), bf16 WMMA, f32 acc
__global__ __launch_bounds__(256) void k_gemm128(
    const float* __restrict__ A, const float* __restrict__ W,
    const float* __restrict__ bias, float* __restrict__ D,
    int n_rows, int do_relu)
{
  __shared__ bf16t sW[HD * SW_STRIDE];     // W transposed: sW[n*SW+k]
  __shared__ bf16t sA[2][HD * SA_STRIDE];  // double-buffered 128x32 A slab

  const int tid  = threadIdx.x;
  const int row0 = blockIdx.x * HD;

  // stage transposed weights once (64KB f32 -> 34KB bf16)
  for (int idx = tid; idx < HD * HD; idx += 256) {
    const int k = idx >> 7;
    const int n = idx & (HD - 1);
    sW[n * SW_STRIDE + k] = (bf16t)W[idx];
  }

  const int lane = tid & 31;
  const int wave = tid >> 5;   // 8 waves, each owns 16 output rows
  const int m    = lane & 15;
  const int half = lane >> 4;

  v8f zero = {};
  v8f acc[8];
#pragma unroll
  for (int t = 0; t < 8; ++t) acc[t] = zero;

  float4 pre[4];
  load_slab(A, row0, n_rows, 0, tid, pre);
  store_slab(sA[0], tid, pre);
  __syncthreads();

#pragma unroll
  for (int kk = 0; kk < 4; ++kk) {
    const int k0 = kk * 32;
    const bf16t* cur = sA[kk & 1];
    bf16t*       nxt = sA[(kk & 1) ^ 1];

    // prefetch next A slab into registers (overlaps HBM latency with WMMA)
    if (kk < 3) load_slab(A, row0, n_rows, k0 + 32, tid, pre);

    // A fragment: lane holds row (lane&15); K runs [8*half,+8) and [16+8*half,+8)
    const bf16t* ap = &cur[(wave * 16 + m) * SA_STRIDE + 8 * half];
    v8bf alo = *(const v8bf*)ap;
    v8bf ahi = *(const v8bf*)(ap + 16);
    v16bf afrag = CAT16(alo, ahi);

#pragma unroll
    for (int t = 0; t < 8; ++t) {
      // B fragment: lane holds col (lane&15); K runs [8*half,+8) and [16+8*half,+8)
      const bf16t* q = &sW[(t * 16 + m) * SW_STRIDE + k0 + 8 * half];
      v8bf blo = *(const v8bf*)q;
      v8bf bhi = *(const v8bf*)(q + 16);
      v16bf bfrag = CAT16(blo, bhi);
      acc[t] = __builtin_amdgcn_wmma_f32_16x16x32_bf16(
          false, afrag, false, bfrag, (short)0, acc[t], false, false);
    }

    if (kk < 3) store_slab(nxt, tid, pre);
    __syncthreads();
  }

  // epilogue: D element (r, lane) -> row r + 8*half, col lane&15
#pragma unroll
  for (int t = 0; t < 8; ++t) {
    const int col = t * 16 + m;
    const float bv = bias[col];
#pragma unroll
    for (int r = 0; r < 8; ++r) {
      const int row = row0 + wave * 16 + r + 8 * half;
      if (row < n_rows) {
        float v = acc[t][r] + bv;
        if (do_relu) v = fmaxf(v, 0.0f);
        D[(size_t)row * HD + col] = v;
      }
    }
  }
}

// --------------------------------------------------------- batchnorm (train)
__global__ __launch_bounds__(128) void k_bnstats(const float* __restrict__ h,
                                                 float* __restrict__ stats, int n_rows) {
  const int c = threadIdx.x;
  float s = 0.0f, s2 = 0.0f;
  for (int row = blockIdx.x; row < n_rows; row += gridDim.x) {
    const float v = h[(size_t)row * HD + c];
    s += v; s2 += v * v;
  }
  atomicAdd(&stats[c], s);
  atomicAdd(&stats[HD + c], s2);
}

__global__ __launch_bounds__(128) void k_bnapply(float* __restrict__ h,
                                                 const float* __restrict__ stats,
                                                 const float* __restrict__ gamma,
                                                 const float* __restrict__ beta, int n_rows) {
  const int c = threadIdx.x;
  const float invn  = 1.0f / (float)n_rows;
  const float mean  = stats[c] * invn;
  const float var   = stats[HD + c] * invn - mean * mean;
  const float scale = gamma[c] * rsqrtf(var + BNEPS);
  const float shift = beta[c] - mean * scale;
  for (int row = blockIdx.x; row < n_rows; row += gridDim.x) {
    const size_t idx = (size_t)row * HD + c;
    h[idx] = h[idx] * scale + shift;
  }
}

// --------------------------------------------------------------- pooling
__global__ void k_counts(const int* __restrict__ batch, float* __restrict__ counts, int n) {
  const int i = blockIdx.x * blockDim.x + threadIdx.x;
  if (i < n) atomicAdd(&counts[batch[i]], 1.0f);
}

// pooled[g][l*128+c] += sum of xs[l][row][c] over rows with batch==g
__global__ __launch_bounds__(128) void k_pool(const float* __restrict__ xs,
                                              const int* __restrict__ batch,
                                              float* __restrict__ pooled, int n_rows) {
  const int l = blockIdx.y;
  const int c = threadIdx.x;
  const float* h = xs + (size_t)l * N_NODES * HD;
  const int row_start = blockIdx.x * 256;
  if (row_start >= n_rows) return;
  const int row_end = min(row_start + 256, n_rows);
  float accv = 0.0f;
  int curg = batch[row_start];
  for (int row = row_start; row < row_end; ++row) {
    const int g = batch[row];
    if (g != curg) {
      atomicAdd(&pooled[(size_t)curg * (NL * HD) + l * HD + c], accv);
      accv = 0.0f; curg = g;
    }
    accv += h[(size_t)row * HD + c];
  }
  atomicAdd(&pooled[(size_t)curg * (NL * HD) + l * HD + c], accv);
}

// --------------------------------------------------------------- head MLP
__global__ __launch_bounds__(128) void k_mlp1(const float* __restrict__ pooled,
                                              const float* __restrict__ counts,
                                              const float* __restrict__ w,
                                              const float* __restrict__ b,
                                              float* __restrict__ hidden) {
  const int g = blockIdx.x;
  const int j = threadIdx.x;
  const float inv = 1.0f / fmaxf(counts[g], 1.0f);
  float s = b[j];
  for (int k = 0; k < NL * HD; ++k)
    s += pooled[(size_t)g * (NL * HD) + k] * inv * w[(size_t)k * HD + j];
  hidden[g * HD + j] = fmaxf(s, 0.0f);
}

__global__ void k_mlp2(const float* __restrict__ hidden, const float* __restrict__ w,
                       const float* __restrict__ b, float* __restrict__ out) {
  const int idx = blockIdx.x * blockDim.x + threadIdx.x;
  if (idx >= NG * NC) return;
  const int g = idx / NC;
  const int c = idx % NC;
  float s = b[c];
  for (int j = 0; j < HD; ++j) s += hidden[g * HD + j] * w[j * NC + c];
  out[idx] = s;
}

// ----------------------------------------------------------------- driver
extern "C" void kernel_launch(void* const* d_in, const int* in_sizes, int n_in,
                              void* d_out, int out_size, void* d_ws, size_t ws_size,
                              hipStream_t stream) {
  (void)in_sizes; (void)n_in; (void)out_size; (void)ws_size;
  const float* x      = (const float*)d_in[0];
  const int*   ei     = (const int*)d_in[1];
  const int*   batch  = (const int*)d_in[2];
  const float* w1_0   = (const float*)d_in[3];
  const float* b1_0   = (const float*)d_in[4];
  const float* w2_0   = (const float*)d_in[5];
  const float* b2_0   = (const float*)d_in[6];
  const float* gamma0 = (const float*)d_in[7];
  const float* beta0  = (const float*)d_in[8];
  const float* w1s    = (const float*)d_in[9];
  const float* b1s    = (const float*)d_in[10];
  const float* w2s    = (const float*)d_in[11];
  const float* b2s    = (const float*)d_in[12];
  const float* gammas = (const float*)d_in[13];
  const float* betas  = (const float*)d_in[14];
  const float* lin1_w = (const float*)d_in[15];
  const float* lin1_b = (const float*)d_in[16];
  const float* lin2_w = (const float*)d_in[17];
  const float* lin2_b = (const float*)d_in[18];
  float* out = (float*)d_out;

  const size_t NH = (size_t)N_NODES * HD;
  float* ws     = (float*)d_ws;
  float* h_in   = ws;                    // N x 128
  float* t1     = ws + NH;               // N x 128
  float* xs     = ws + 2 * NH;           // 4 x N x 128 (JK concat slices)
  float* stats  = ws + 6 * NH;           // 256
  float* pooled = stats + 2 * HD;        // G x 512
  float* counts = pooled + (size_t)NG * NL * HD;  // G
  float* hidden = counts + NG;           // G x 128

  const int gemmGrid = (N_NODES + HD - 1) / HD;

  k_zero<<<64, 256, 0, stream>>>(pooled, NG * NL * HD + NG);
  k_counts<<<(N_NODES + 255) / 256, 256, 0, stream>>>(batch, counts, N_NODES);

  for (int l = 0; l < NL; ++l) {
    const float* in = (l == 0) ? x      : xs + (size_t)(l - 1) * NH;
    const float* w1 = (l == 0) ? w1_0   : w1s + (size_t)(l - 1) * HD * HD;
    const float* b1 = (l == 0) ? b1_0   : b1s + (size_t)(l - 1) * HD;
    const float* w2 = (l == 0) ? w2_0   : w2s + (size_t)(l - 1) * HD * HD;
    const float* b2 = (l == 0) ? b2_0   : b2s + (size_t)(l - 1) * HD;
    const float* ga = (l == 0) ? gamma0 : gammas + (size_t)(l - 1) * HD;
    const float* be = (l == 0) ? beta0  : betas + (size_t)(l - 1) * HD;
    float* hl = xs + (size_t)l * NH;

    k_copy4<<<2048, 256, 0, stream>>>((float4*)h_in, (const float4*)in, NH / 4);
    k_edge<<<8192, 256, 0, stream>>>(in, ei, h_in);
    k_gemm128<<<gemmGrid, 256, 0, stream>>>(h_in, w1, b1, t1, N_NODES, 1);
    k_gemm128<<<gemmGrid, 256, 0, stream>>>(t1, w2, b2, hl, N_NODES, 1);
    k_zero<<<1, 256, 0, stream>>>(stats, 2 * HD);
    k_bnstats<<<512, HD, 0, stream>>>(hl, stats, N_NODES);
    k_bnapply<<<1024, HD, 0, stream>>>(hl, stats, ga, be, N_NODES);
  }

  dim3 pgrid((N_NODES + 255) / 256, NL);
  k_pool<<<pgrid, HD, 0, stream>>>(xs, batch, pooled, N_NODES);
  k_mlp1<<<NG, HD, 0, stream>>>(pooled, counts, lin1_w, lin1_b, hidden);
  k_mlp2<<<(NG * NC + 255) / 256, 256, 0, stream>>>(hidden, lin2_w, lin2_b, out);
}
